// NATBlock_57535381897318
// MI455X (gfx1250) — matrix operand order repaired and checked by reading, not compiled, not run on hardware
//
#include <hip/hip_runtime.h>
#include <hip/hip_bf16.h>
#include <math.h>

// ---------------- problem constants ----------------
#define BDIM   4
#define CDIM   512
#define HDIM   28
#define WDIM   28
#define HWDIM  (HDIM * WDIM)        // 784
#define MDIM   (BDIM * HWDIM)       // 3136
#define NHEADS 16
#define DH     32
#define KW     7
#define EPSBN  1e-5f

typedef __bf16 bf16_t;
typedef bf16_t bf16x8 __attribute__((ext_vector_type(8)));
typedef bf16_t v16bf  __attribute__((ext_vector_type(16)));
typedef float  v8f    __attribute__((ext_vector_type(8)));

union FragAB { v16bf v; bf16x8 h[2]; };

// low 32 bits of a generic pointer to __shared__ = wave-relative LDS byte offset
// (ISA 10.2: LDS aperture addresses truncate to addr[31:0])
__device__ __forceinline__ unsigned lds_off_of(const void* p) {
    return (unsigned)(size_t)p;
}

// ---------------- layout / convert kernels ----------------
__global__ __launch_bounds__(256) void k_f32_to_bf16(const float* __restrict__ src,
                                                     bf16_t* __restrict__ dst, int n) {
    int i = blockIdx.x * 256 + threadIdx.x;
    if (i < n) dst[i] = (bf16_t)src[i];
}

// x (B,C,H,W) fp32 -> xh (M, C) bf16  (pixel-major, channel contiguous)
__global__ __launch_bounds__(256) void k_transpose_x(const float* __restrict__ x,
                                                     bf16_t* __restrict__ xh) {
    int i = blockIdx.x * 256 + threadIdx.x;
    if (i >= MDIM * CDIM) return;
    int c = i & (CDIM - 1);
    int m = i >> 9;                 // /CDIM
    int b = m / HWDIM;
    int pix = m - b * HWDIM;
    xh[i] = (bf16_t)x[((size_t)b * CDIM + c) * HWDIM + pix];
}

// y (M,C) fp32 -> x1 (M,C) bf16 with per-channel affine (folded batchnorm)
__global__ __launch_bounds__(256) void k_normalize_bf16(const float* __restrict__ y,
                                                        const float* __restrict__ sc,
                                                        const float* __restrict__ sh,
                                                        bf16_t* __restrict__ out, int n) {
    int i = blockIdx.x * 256 + threadIdx.x;
    if (i >= n) return;
    int c = i & (CDIM - 1);
    out[i] = (bf16_t)(y[i] * sc[c] + sh[c]);
}

// ---------------- batchnorm statistics -> channel affine ----------------
__global__ __launch_bounds__(256) void k_bn_stats(const float* __restrict__ y,
                                                  const float* __restrict__ g,
                                                  const float* __restrict__ be,
                                                  float* __restrict__ scale,
                                                  float* __restrict__ shift) {
    __shared__ float ss[256], sq[256];
    int c = blockIdx.x;
    int t = threadIdx.x;
    float s = 0.f, q = 0.f;
    for (int m = t; m < MDIM; m += 256) {
        float v = y[(size_t)m * CDIM + c];
        s += v; q += v * v;
    }
    ss[t] = s; sq[t] = q;
    __syncthreads();
    for (int o = 128; o > 0; o >>= 1) {
        if (t < o) { ss[t] += ss[t + o]; sq[t] += sq[t + o]; }
        __syncthreads();
    }
    if (t == 0) {
        float mean = ss[0] * (1.0f / MDIM);
        float var  = sq[0] * (1.0f / MDIM) - mean * mean;
        float rstd = rsqrtf(var + EPSBN);
        float scv  = g[c] * rstd;
        scale[c] = scv;
        shift[c] = be[c] - mean * scv;
    }
}

// ---------------- epilogues (per-column prep hoists n-indexed loads) ----------------
struct EpiQKV {       // qkv = A*W^T + b  -> fp32 (M,1536)
    const float* __restrict__ bias; float* __restrict__ out;
    struct Col { float b; };
    __device__ Col prep(int n) const { return {bias[n]}; }
    __device__ void store(float v, int m, int n, const Col& c) const {
        out[(size_t)m * 1536 + n] = v + c.b;
    }
};
struct EpiProj {      // y1 = x + (A*W^T + b)  -> fp32 (M,C)
    const float* __restrict__ bias; const float* __restrict__ x; float* __restrict__ y1;
    struct Col { float b; };
    __device__ Col prep(int n) const { return {bias[n]}; }
    __device__ void store(float v, int m, int n, const Col& c) const {
        int b_ = m / HWDIM, pix = m - b_ * HWDIM;
        float xv = x[((size_t)b_ * CDIM + n) * HWDIM + pix];
        y1[(size_t)m * CDIM + n] = xv + v + c.b;
    }
};
struct EpiMLP1 {      // t = gelu(A*W1^T + b1) -> bf16 (M,1024)
    const float* __restrict__ bias; bf16_t* __restrict__ t;
    struct Col { float b; };
    __device__ Col prep(int n) const { return {bias[n]}; }
    __device__ void store(float v, int m, int n, const Col& c) const {
        float u = v + c.b;
        float gl = 0.5f * u * (1.0f + erff(u * 0.70710678118654752f));
        t[(size_t)m * 1024 + n] = (bf16_t)gl;
    }
};
struct EpiMLP2 {      // y2 = x1 + (A*W2^T + b2) -> fp32 (M,C); x1 recomputed from y1
    const float* __restrict__ bias; const float* __restrict__ y1;
    const float* __restrict__ sc1; const float* __restrict__ sh1; float* __restrict__ y2;
    struct Col { float b, sc, sh; };
    __device__ Col prep(int n) const { return {bias[n], sc1[n], sh1[n]}; }
    __device__ void store(float v, int m, int n, const Col& c) const {
        float x1 = y1[(size_t)m * CDIM + n] * c.sc + c.sh;
        y2[(size_t)m * CDIM + n] = x1 + v + c.b;
    }
};

// ---------------- bf16 WMMA GEMM:  C(M,N) = A(M,K) * W(N,K)^T ----------------
// block: 256 threads (8 waves), tile 64M x 128N, BK=32, double-buffered LDS.
// Tile staging: CDNA5 async global->LDS DMA (ASYNCcnt), overlapped with WMMA.
// wave (wm,wn) in 2x4 grid computes 32x32 via 2x2 v_wmma_f32_16x16x32_bf16.
template <class Epi>
__global__ __launch_bounds__(256) void k_gemm_bf16(const bf16_t* __restrict__ A,
                                                   const bf16_t* __restrict__ Bw,
                                                   int M, int N, int K, Epi epi) {
    constexpr int BM = 64, BN = 128, BK = 32, LD = 40; // LD padded: 80B stride, conflict-free
    __shared__ bf16_t As[2][BM * LD];
    __shared__ bf16_t Bs[2][BN * LD];

    const int tid  = threadIdx.x;
    const int lane = tid & 31;
    const int wave = tid >> 5;
    const int wm   = wave >> 2;   // 0..1
    const int wn   = wave & 3;    // 0..3
    const int m0   = blockIdx.y * BM;
    const int n0   = blockIdx.x * BN;

    // per-thread staging addresses
    const int arow = tid >> 2, acg = tid & 3;      // A: 64 rows x 4 chunks of 8 bf16
    const int brow = tid >> 1, bcg = tid & 1;      // B: 128 rows x 2 chunks of 16 bf16
    const unsigned a_lds0 = lds_off_of(&As[0][arow * LD + acg * 8]);
    const unsigned a_lds1 = lds_off_of(&As[1][arow * LD + acg * 8]);
    const unsigned b_lds0 = lds_off_of(&Bs[0][brow * LD + bcg * 16]);
    const unsigned b_lds1 = lds_off_of(&Bs[1][brow * LD + bcg * 16]);
    const bf16_t* a_g = A + (size_t)(m0 + arow) * K + acg * 8;
    const bf16_t* b_g = Bw + (size_t)(n0 + brow) * K + bcg * 16;

    auto issue = [&](int buf, int k0) {
        const bf16_t* ag = a_g + k0;
        const bf16_t* bg = b_g + k0;
        unsigned al = buf ? a_lds1 : a_lds0;
        unsigned bl = buf ? b_lds1 : b_lds0;
        asm volatile("global_load_async_to_lds_b128 %0, %1, off"
                     :: "v"(al), "v"(ag) : "memory");
        asm volatile("global_load_async_to_lds_b128 %0, %1, off"
                     :: "v"(bl), "v"(bg) : "memory");
        asm volatile("global_load_async_to_lds_b128 %0, %1, off offset:16"
                     :: "v"(bl + 16u), "v"(bg) : "memory");
    };

    const v8f vzero = {0.f,0.f,0.f,0.f,0.f,0.f,0.f,0.f};
    v8f acc[2][2];
#pragma unroll
    for (int i = 0; i < 2; i++)
#pragma unroll
        for (int j = 0; j < 2; j++) acc[i][j] = vzero;

    issue(0, 0);
    int buf = 0;
    for (int k0 = 0; k0 < K; k0 += BK, buf ^= 1) {
        // current buffer's DMA complete (own wave), then block-wide visibility
        asm volatile("s_wait_asynccnt 0" ::: "memory");
        __syncthreads();
        // prefetch next tile while computing this one
        if (k0 + BK < K) issue(buf ^ 1, k0 + BK);

        const bf16_t* asb = &As[buf][0];
        const bf16_t* bsb = &Bs[buf][0];

        // A fragment (16-bit 16x32): lanes 0-15 hold K {0..7,16..23}, lanes 16-31 {8..15,24..31}
        FragAB a[2], b[2];
#pragma unroll
        for (int t = 0; t < 2; t++) {
            int row = wm * 32 + t * 16 + (lane & 15);
            int off = (lane >> 4) * 8;
            a[t].h[0] = *reinterpret_cast<const bf16x8*>(&asb[row * LD + off]);
            a[t].h[1] = *reinterpret_cast<const bf16x8*>(&asb[row * LD + off + 16]);
        }
        // B fragment (16-bit 32x16): lane n = column, lanes 0-15 K {0..7,16..23}, lanes 16-31 {8..15,24..31}
#pragma unroll
        for (int t = 0; t < 2; t++) {
            int col = wn * 32 + t * 16 + (lane & 15);
            int off = (lane >> 4) * 8;
            b[t].h[0] = *reinterpret_cast<const bf16x8*>(&bsb[col * LD + off]);
            b[t].h[1] = *reinterpret_cast<const bf16x8*>(&bsb[col * LD + off + 16]);
        }

#pragma unroll
        for (int i = 0; i < 2; i++)
#pragma unroll
            for (int j = 0; j < 2; j++)
                acc[i][j] = __builtin_amdgcn_wmma_f32_16x16x32_bf16(
                    false, a[i].v, false, b[j].v, (short)0, acc[i][j], false, false);
        __syncthreads();   // all waves done reading buf before it is overwritten
    }

    // C/D layout: VGPR r -> M=r (lanes 0-15) / M=8+r (lanes 16-31); N = lane&15
#pragma unroll
    for (int j = 0; j < 2; j++) {
        const int n = n0 + wn * 32 + j * 16 + (lane & 15);
        const auto col = epi.prep(n);
#pragma unroll
        for (int i = 0; i < 2; i++)
#pragma unroll
            for (int r = 0; r < 8; r++) {
                int m = m0 + wm * 32 + i * 16 + ((lane >> 4) * 8) + r;
                epi.store(acc[i][j][r], m, n, col);
            }
    }
}

// ---------------- neighborhood attention ----------------
// one block per (b, head, row i); 28 waves = one per pixel column j; lane = d (DH=32)
__global__ __launch_bounds__(896) void k_natt(const float* __restrict__ qkv,
                                              const float* __restrict__ rpb,
                                              bf16_t* __restrict__ out) {
    __shared__ float sk[KW * WDIM * DH];   // 25 KB
    __shared__ float sv[KW * WDIM * DH];   // 25 KB

    const int blk  = blockIdx.x;
    const int i    = blk % HDIM;
    const int head = (blk / HDIM) % NHEADS;
    const int b    = blk / (HDIM * NHEADS);

    int start_i = i - (KW / 2);
    if (start_i < 0) start_i = 0;
    if (start_i > HDIM - KW) start_i = HDIM - KW;

    const int tid = threadIdx.x;
    // stage K/V for the 7 neighbor rows of this image row
    for (int e = tid; e < KW * WDIM * DH; e += 896) {
        int d   = e & 31;
        int col = (e >> 5) % WDIM;
        int kr  = (e >> 5) / WDIM;
        int row = start_i + kr;
        size_t base = (((size_t)(b * HDIM + row) * WDIM + col) * 1536) + head * DH + d;
        sk[e] = qkv[base + 512];
        sv[e] = qkv[base + 1024];
    }
    __syncthreads();

    const int j = tid >> 5;     // pixel column 0..27
    const int d = tid & 31;
    int start_j = j - (KW / 2);
    if (start_j < 0) start_j = 0;
    if (start_j > WDIM - KW) start_j = WDIM - KW;

    const size_t mj = (size_t)(b * HDIM + i) * WDIM + j;
    const float scale = 0.17677669529663687f;   // 32^-0.5
    float q = qkv[mj * 1536 + head * DH + d] * scale;

    float lg[KW * KW];
    float lmax = -1e30f;
#pragma unroll
    for (int kr = 0; kr < KW; kr++) {
        int relr = (start_i + kr) - i + (KW - 1);
#pragma unroll
        for (int kc = 0; kc < KW; kc++) {
            int col = start_j + kc;
            float p = q * sk[((kr * WDIM) + col) * DH + d];
            // wave32 all-reduce over d
            for (int o = 16; o > 0; o >>= 1) p += __shfl_xor(p, o, 32);
            int relc = col - j + (KW - 1);
            float l = p + rpb[((size_t)head * 13 + relr) * 13 + relc];
            lg[kr * KW + kc] = l;
            lmax = fmaxf(lmax, l);
        }
    }

    float denom = 0.f, o_d = 0.f;
#pragma unroll
    for (int kr = 0; kr < KW; kr++)
#pragma unroll
        for (int kc = 0; kc < KW; kc++) {
            int col = start_j + kc;
            float e = __expf(lg[kr * KW + kc] - lmax);
            denom += e;
            o_d += e * sv[((kr * WDIM) + col) * DH + d];
        }
    out[mj * CDIM + head * DH + d] = (bf16_t)(o_d / denom);
}

// ---------------- final write-back (BHWC fp32 -> BCHW fp32, folded BN2) ----------------
__global__ __launch_bounds__(256) void k_final(const float* __restrict__ y2,
                                               const float* __restrict__ sc,
                                               const float* __restrict__ sh,
                                               float* __restrict__ out) {
    int idx = blockIdx.x * 256 + threadIdx.x;
    if (idx >= BDIM * CDIM * HWDIM) return;
    int pix = idx % HWDIM;
    int c   = (idx / HWDIM) & (CDIM - 1);
    int b   = idx / (CDIM * HWDIM);
    int m   = b * HWDIM + pix;
    out[idx] = y2[(size_t)m * CDIM + c] * sc[c] + sh[c];
}

// ---------------- launcher ----------------
extern "C" void kernel_launch(void* const* d_in, const int* in_sizes, int n_in,
                              void* d_out, int out_size, void* d_ws, size_t ws_size,
                              hipStream_t stream) {
    const float* x      = (const float*)d_in[0];
    const float* qkv_w  = (const float*)d_in[1];
    const float* qkv_b  = (const float*)d_in[2];
    const float* proj_w = (const float*)d_in[3];
    const float* proj_b = (const float*)d_in[4];
    const float* rpb    = (const float*)d_in[5];
    const float* w1     = (const float*)d_in[6];
    const float* b1     = (const float*)d_in[7];
    const float* w2     = (const float*)d_in[8];
    const float* b2     = (const float*)d_in[9];
    const float* g1     = (const float*)d_in[10];
    const float* be1    = (const float*)d_in[11];
    const float* g2     = (const float*)d_in[12];
    const float* be2    = (const float*)d_in[13];

    char* ws = (char*)d_ws;
    size_t off = 0;
    auto alloc = [&](size_t bytes) -> char* {
        char* p = ws + off;
        off += (bytes + 255) & ~size_t(255);
        return p;
    };
    bf16_t* xh     = (bf16_t*)alloc((size_t)MDIM * CDIM * 2);
    bf16_t* qkvw_b = (bf16_t*)alloc((size_t)3 * CDIM * CDIM * 2);
    bf16_t* projw_b= (bf16_t*)alloc((size_t)CDIM * CDIM * 2);
    bf16_t* w1_b   = (bf16_t*)alloc((size_t)2 * CDIM * CDIM * 2);
    bf16_t* w2_b   = (bf16_t*)alloc((size_t)2 * CDIM * CDIM * 2);
    float*  qkvbuf = (float*)alloc((size_t)MDIM * 3 * CDIM * 4);
    bf16_t* attn_b = (bf16_t*)alloc((size_t)MDIM * CDIM * 2);
    float*  y1     = (float*)alloc((size_t)MDIM * CDIM * 4);
    bf16_t* x1_b   = (bf16_t*)alloc((size_t)MDIM * CDIM * 2);
    bf16_t* tbuf   = (bf16_t*)alloc((size_t)MDIM * 2 * CDIM * 2);
    float*  y2     = (float*)alloc((size_t)MDIM * CDIM * 4);
    float*  sc1    = (float*)alloc(CDIM * 4);
    float*  sh1    = (float*)alloc(CDIM * 4);
    float*  sc2    = (float*)alloc(CDIM * 4);
    float*  sh2    = (float*)alloc(CDIM * 4);
    (void)ws_size; (void)in_sizes; (void)n_in; (void)out_size;

    // 1) layout conversion
    k_transpose_x<<<(MDIM * CDIM + 255) / 256, 256, 0, stream>>>(x, xh);
    k_f32_to_bf16<<<(3 * CDIM * CDIM + 255) / 256, 256, 0, stream>>>(qkv_w, qkvw_b, 3 * CDIM * CDIM);
    k_f32_to_bf16<<<(CDIM * CDIM + 255) / 256, 256, 0, stream>>>(proj_w, projw_b, CDIM * CDIM);
    k_f32_to_bf16<<<(2 * CDIM * CDIM + 255) / 256, 256, 0, stream>>>(w1, w1_b, 2 * CDIM * CDIM);
    k_f32_to_bf16<<<(2 * CDIM * CDIM + 255) / 256, 256, 0, stream>>>(w2, w2_b, 2 * CDIM * CDIM);

    // 2) QKV GEMM: (3136 x 1536) = xh (3136x512) * qkv_w^T
    k_gemm_bf16<EpiQKV><<<dim3(1536 / 128, MDIM / 64), 256, 0, stream>>>(
        xh, qkvw_b, MDIM, 1536, CDIM, EpiQKV{qkv_b, qkvbuf});

    // 3) neighborhood attention -> bf16 (M, C)
    k_natt<<<BDIM * NHEADS * HDIM, 896, 0, stream>>>(qkvbuf, rpb, attn_b);

    // 4) proj GEMM + residual: y1 = x + attn*proj_w^T + b
    k_gemm_bf16<EpiProj><<<dim3(CDIM / 128, MDIM / 64), 256, 0, stream>>>(
        attn_b, projw_b, MDIM, CDIM, CDIM, EpiProj{proj_b, x, y1});

    // 5) BN1 stats -> channel affine; materialize x1 as bf16
    k_bn_stats<<<CDIM, 256, 0, stream>>>(y1, g1, be1, sc1, sh1);
    k_normalize_bf16<<<(MDIM * CDIM + 255) / 256, 256, 0, stream>>>(y1, sc1, sh1, x1_b, MDIM * CDIM);

    // 6) MLP1: t = gelu(x1 * w1^T + b1)  (bf16 out, feeds MLP2)
    k_gemm_bf16<EpiMLP1><<<dim3(1024 / 128, MDIM / 64), 256, 0, stream>>>(
        x1_b, w1_b, MDIM, 1024, CDIM, EpiMLP1{b1, tbuf});

    // 7) MLP2 + residual: y2 = x1 + t * w2^T + b2
    k_gemm_bf16<EpiMLP2><<<dim3(CDIM / 128, MDIM / 64), 256, 0, stream>>>(
        tbuf, w2_b, MDIM, CDIM, 2 * CDIM, EpiMLP2{b2, y1, sc1, sh1, y2});

    // 8) BN2 stats + final BCHW write
    k_bn_stats<<<CDIM, 256, 0, stream>>>(y2, g2, be2, sc2, sh2);
    k_final<<<(BDIM * CDIM * HWDIM + 255) / 256, 256, 0, stream>>>(y2, sc2, sh2, (float*)d_out);
}
